// Net_56212531970582
// MI455X (gfx1250) — matrix operand order repaired and checked by reading, not compiled, not run on hardware
//
#include <hip/hip_runtime.h>

// ---------------------------------------------------------------------------
// RGCN (basis-decomposed, mean aggregation) + decoder for MI455X (gfx1250).
// Aggregate-before-transform: per-edge work is a cheap fp32 atomic scatter in
// the *input* space, all dense math runs through v_wmma_f32_16x16x32_bf16
// (fp32 -> bf16 via v_cvt_pk_bf16_f32 in registers, fp32 accumulation).
// ---------------------------------------------------------------------------

typedef __attribute__((ext_vector_type(16))) __bf16 v16bf_t;
typedef __attribute__((ext_vector_type(8)))  float  v8f_t;
typedef __attribute__((ext_vector_type(4)))  float  f4_t;

// ---------------------------------------------------------------------------
// Tiled WMMA GEMM, N known at compile time (N % (NT*16) == 0, M % 16 == 0).
// One wave computes a 16 x (NT*16) strip: one A fragment reused over NT
// accumulators (NT=8 => one pass over N=128, so A is read exactly once).
// Main K loop is branch-free (unconditional loads, immediate offsets);
// K tail (K % 32) uses clamped addresses + selects, no exec churn.
// NTA: stream A with non-temporal hint (for the 400MB x matrix, read once).
// ---------------------------------------------------------------------------
template <int NDIM, int NT, bool NTA>
__global__ __launch_bounds__(256)
void wmma_gemm_kernel(const float* __restrict__ A, const float* __restrict__ B,
                      float* __restrict__ C, int M, int K,
                      const float* __restrict__ bias, int do_relu) {
    constexpr int STRIP = NT * 16;
    static_assert(NDIM % STRIP == 0, "N must be a multiple of the strip");

    const int lane = threadIdx.x & 31;
    const int wave = threadIdx.x >> 5;
    const int wavesPerBlock = blockDim.x >> 5;

    const int tilesM = M >> 4;
    constexpr int tilesN = NDIM / STRIP;
    const int tile = blockIdx.x * wavesPerBlock + wave;
    if (tile >= tilesM * tilesN) return;                   // wave-uniform guard

    const int tm = tile / tilesN;
    const int tn = tile % tilesN;
    const int m0 = tm << 4;
    const int n0 = tn * STRIP;

    const int hi = lane >> 4;                              // lane half (0/1)
    const int lm = lane & 15;

    // A fragment layout (16x32 bf16):
    //  lane<16 : elems 0..7 -> K=k0+0..7,   elems 8..15 -> K=k0+16..23
    //  lane>=16: elems 0..7 -> K=k0+8..15,  elems 8..15 -> K=k0+24..31
    const float* Arow = A + (size_t)(m0 + lm) * K + (hi << 3);
    // B fragment layout (32x16 bf16): col = lane%16, K = k0 + 8*hi + (e&7) + 16*(e>>3)
    const float* Bp = B + (size_t)(hi << 3) * NDIM + n0 + lm;

    v8f_t acc[NT];
#pragma unroll
    for (int nt = 0; nt < NT; ++nt) acc[nt] = (v8f_t){};

    const int kMain = K & ~31;
    for (int k0 = 0; k0 < kMain; k0 += 32) {
        // ---- A: two contiguous 8-float runs -> 4 aligned 128b loads -------
        f4_t a0, a1, a2, a3;
        if (NTA) {
            a0 = __builtin_nontemporal_load((const f4_t*)(Arow + k0));
            a1 = __builtin_nontemporal_load((const f4_t*)(Arow + k0 + 4));
            a2 = __builtin_nontemporal_load((const f4_t*)(Arow + k0 + 16));
            a3 = __builtin_nontemporal_load((const f4_t*)(Arow + k0 + 20));
        } else {
            a0 = *(const f4_t*)(Arow + k0);
            a1 = *(const f4_t*)(Arow + k0 + 4);
            a2 = *(const f4_t*)(Arow + k0 + 16);
            a3 = *(const f4_t*)(Arow + k0 + 20);
        }
        v16bf_t a;
#pragma unroll
        for (int e = 0; e < 4; ++e) {
            a[e]      = (__bf16)a0[e];
            a[e + 4]  = (__bf16)a1[e];
            a[e + 8]  = (__bf16)a2[e];
            a[e + 12] = (__bf16)a3[e];
        }

        // ---- B: NT fragments, compile-time immediate offsets --------------
        const float* Bk = Bp + (size_t)k0 * NDIM;
        v16bf_t b[NT];
#pragma unroll
        for (int nt = 0; nt < NT; ++nt)
#pragma unroll
            for (int e = 0; e < 16; ++e)
                b[nt][e] = (__bf16)Bk[((e & 7) + ((e >> 3) << 4)) * NDIM + nt * 16];

#pragma unroll
        for (int nt = 0; nt < NT; ++nt)
            acc[nt] = __builtin_amdgcn_wmma_f32_16x16x32_bf16(
                false, a, false, b[nt], (short)0, acc[nt], false, false);
    }

    if (kMain < K) {                                       // K tail (< 32)
        v16bf_t a;
#pragma unroll
        for (int e = 0; e < 16; ++e) {
            const int ka = kMain + (e & 7) + ((e >> 3) << 4);  // +8*hi via Arow
            const int kc = (ka < K - (hi << 3)) ? ka : 0;      // clamp in-bounds
            float av = Arow[kc];
            if (ka + (hi << 3) >= K) av = 0.0f;
            a[e] = (__bf16)av;
        }
        const float* Bk = Bp + (size_t)kMain * NDIM;
        v16bf_t b[NT];
#pragma unroll
        for (int nt = 0; nt < NT; ++nt)
#pragma unroll
            for (int e = 0; e < 16; ++e) {
                const int eo = (e & 7) + ((e >> 3) << 4);
                const int kb = kMain + (hi << 3) + eo;
                float bv = Bk[(kb < K ? eo : -(hi << 3) - eo) * NDIM + nt * 16];
                if (kb >= K) bv = 0.0f;
                b[nt][e] = (__bf16)bv;
            }
#pragma unroll
        for (int nt = 0; nt < NT; ++nt)
            acc[nt] = __builtin_amdgcn_wmma_f32_16x16x32_bf16(
                false, a, false, b[nt], (short)0, acc[nt], false, false);
    }

    // ---- store: C layout = VGPR j -> row m0+j (+8 upper half), col=lane%16
#pragma unroll
    for (int nt = 0; nt < NT; ++nt) {
        const int col = n0 + (nt << 4) + lm;
        const float bv = bias ? bias[col] : 0.0f;
        float* Cp = C + (size_t)(m0 + (hi << 3)) * NDIM + col;
#pragma unroll
        for (int j = 0; j < 8; ++j) {
            float v = acc[nt][j] + bv;
            if (do_relu) v = fmaxf(v, 0.0f);
            Cp[(size_t)j * NDIM] = v;
        }
    }
}

// ---------------------------------------------------------------------------
// Edge degree count (mean normalization denominator)
// ---------------------------------------------------------------------------
__global__ void deg_kernel(const int* __restrict__ dst, float* __restrict__ deg, int E) {
    const int e = blockIdx.x * blockDim.x + threadIdx.x;
    if (e < E) atomicAdd(deg + dst[e], 1.0f);
}

// ---------------------------------------------------------------------------
// Edge scatter: sum_h[dst, rel, :] += h[src, :]   (one wave per edge)
// ---------------------------------------------------------------------------
__global__ void scatter_kernel(const float* __restrict__ h, const int* __restrict__ src,
                               const int* __restrict__ dst, const int* __restrict__ et,
                               float* __restrict__ sumh, int E, int D, int R) {
    const int e    = (int)((blockIdx.x * (size_t)blockDim.x + threadIdx.x) >> 5);
    const int lane = threadIdx.x & 31;
    if (e >= E) return;
    const int s = src[e];
    const int d = dst[e];
    const int r = et[e];
    const float* hs = h + (size_t)s * D;
    float* out = sumh + ((size_t)d * R + r) * D;
    for (int i = lane; i < D; i += 32) atomicAdd(out + i, hs[i]);
}

// ---------------------------------------------------------------------------
// Fold comp into basis: Wcat[r*Din + i, :] = sum_b comp[r,b] * basis[b, i, :]
// ---------------------------------------------------------------------------
__global__ void build_w_kernel(const float* __restrict__ comp, const float* __restrict__ basis,
                               float* __restrict__ W, int R, int IO) {
    const int i = blockIdx.x * blockDim.x + threadIdx.x;
    if (i >= R * IO) return;
    const int r = i / IO;
    const int j = i % IO;
    float acc = 0.0f;
    for (int b = 0; b < R; ++b) acc += comp[r * R + b] * basis[(size_t)b * IO + j];
    W[i] = acc;
}

// ---------------------------------------------------------------------------
// RGCN epilogue: out = relu(agg/max(deg,1) + root_c + bias)
// ---------------------------------------------------------------------------
__global__ void rgcn_epilogue_kernel(const float* __restrict__ agg, const float* __restrict__ rootc,
                                     const float* __restrict__ bias, const float* __restrict__ deg,
                                     float* __restrict__ out, int rows, int D) {
    const int i = blockIdx.x * blockDim.x + threadIdx.x;
    if (i >= rows * D) return;
    const int row = i / D;
    const int col = i - row * D;
    const float dg = fmaxf(deg[row], 1.0f);
    const float v = agg[i] / dg + rootc[i] + bias[col];
    out[i] = fmaxf(v, 0.0f);
}

// ---------------------------------------------------------------------------
// Softmax over 32 classes: one wave per row, lane == class (wave32-exact).
// ---------------------------------------------------------------------------
__global__ void softmax32_kernel(const float* __restrict__ logits, float* __restrict__ out, int rows) {
    const int row  = (int)((blockIdx.x * (size_t)blockDim.x + threadIdx.x) >> 5);
    const int lane = threadIdx.x & 31;
    if (row >= rows) return;
    const float v = logits[(size_t)row * 32 + lane];
    float m = v;
#pragma unroll
    for (int o = 16; o > 0; o >>= 1) m = fmaxf(m, __shfl_xor(m, o, 32));
    const float e = __expf(v - m);
    float s = e;
#pragma unroll
    for (int o = 16; o > 0; o >>= 1) s += __shfl_xor(s, o, 32);
    out[(size_t)row * 32 + lane] = e / s;
}

// ---------------------------------------------------------------------------
// Host-side orchestration (graph-capture safe: only kernel launches and
// hipMemsetAsync on `stream`). Workspace requirement: ~178 MB.
// ---------------------------------------------------------------------------
extern "C" void kernel_launch(void* const* d_in, const int* in_sizes, int n_in,
                              void* d_out, int out_size, void* d_ws, size_t ws_size,
                              hipStream_t stream) {
    const float* x      = (const float*)d_in[0];
    const int*   eidx   = (const int*)  d_in[1];
    const int*   etype  = (const int*)  d_in[2];
    const float* emb    = (const float*)d_in[3];
    const float* comp1  = (const float*)d_in[4];
    const float* basis1 = (const float*)d_in[5];
    const float* root1  = (const float*)d_in[6];
    const float* bias1  = (const float*)d_in[7];
    const float* comp2  = (const float*)d_in[8];
    const float* basis2 = (const float*)d_in[9];
    const float* root2  = (const float*)d_in[10];
    const float* bias2  = (const float*)d_in[11];
    const float* w_dec  = (const float*)d_in[12];

    const int Nn = 10000, IN = 128, HID = 256, EMBD = 128, R = 8, C = 32;
    const int E = in_sizes[2];                 // 640000
    const int* src = eidx;
    const int* dst = eidx + E;

    float* ws = (float*)d_ws;
    size_t off = 0;
    auto alloc = [&](size_t n) { float* p = ws + off; off += n; return p; };

    float* h0     = alloc((size_t)Nn * IN);
    float* h1     = alloc((size_t)Nn * HID);
    float* h2     = alloc((size_t)Nn * EMBD);
    float* sum1   = alloc((size_t)Nn * R * IN);      // [N, R, 128]
    float* sum2   = alloc((size_t)Nn * R * HID);     // [N, R, 256]
    float* deg    = alloc((size_t)Nn);
    float* W1     = alloc((size_t)R * IN * HID);     // [R*128, 256]
    float* W2     = alloc((size_t)R * HID * EMBD);   // [R*256, 128]
    float* agg1   = alloc((size_t)Nn * HID);
    float* root1c = alloc((size_t)Nn * HID);
    float* agg2   = alloc((size_t)Nn * EMBD);
    float* root2c = alloc((size_t)Nn * EMBD);
    float* logits = alloc((size_t)Nn * C);
    (void)ws_size; (void)n_in; (void)out_size;

    auto cdiv = [](int a, int b) { return (a + b - 1) / b; };
    // 8 waves per 256-thread block; strip = NT*16 columns per wave
    auto gemm_blocks = [&](int M, int N, int NT) {
        const int tiles = (M / 16) * (N / (NT * 16));
        return cdiv(tiles, 8);
    };

    // ---- zero accumulation buffers (graph-legal memsets) ------------------
    hipMemsetAsync(sum1, 0, (size_t)Nn * R * IN  * sizeof(float), stream);
    hipMemsetAsync(sum2, 0, (size_t)Nn * R * HID * sizeof(float), stream);
    hipMemsetAsync(deg,  0, (size_t)Nn * sizeof(float), stream);

    // ---- h0 = relu(x @ emb) : [10000,10000]@[10000,128] -------------------
    // NT=8 => full-N strip: x (400MB) streamed from HBM exactly once, with
    // non-temporal hint so it doesn't evict the hot 5MB emb panel from L2.
    wmma_gemm_kernel<128, 8, true><<<gemm_blocks(Nn, 128, 8), 256, 0, stream>>>(
        x, emb, h0, Nn, Nn, nullptr, 1);

    // ---- degrees + layer-1 scatter ---------------------------------------
    deg_kernel<<<cdiv(E, 256), 256, 0, stream>>>(dst, deg, E);
    scatter_kernel<<<cdiv(E * 8, 256), 256, 0, stream>>>(h0, src, dst, etype,
                                                         sum1, E, IN, R);

    // ---- W1 = comp1 . basis1 ; agg1 = sum1 @ W1 ; root1c = h0 @ root1 ----
    build_w_kernel<<<cdiv(R * IN * HID, 256), 256, 0, stream>>>(comp1, basis1, W1,
                                                                R, IN * HID);
    wmma_gemm_kernel<256, 8, false><<<gemm_blocks(Nn, 256, 8), 256, 0, stream>>>(
        sum1, W1, agg1, Nn, R * IN, nullptr, 0);
    wmma_gemm_kernel<256, 8, false><<<gemm_blocks(Nn, 256, 8), 256, 0, stream>>>(
        h0, root1, root1c, Nn, IN, nullptr, 0);
    rgcn_epilogue_kernel<<<cdiv(Nn * HID, 256), 256, 0, stream>>>(
        agg1, root1c, bias1, deg, h1, Nn, HID);

    // ---- layer 2 ----------------------------------------------------------
    scatter_kernel<<<cdiv(E * 8, 256), 256, 0, stream>>>(h1, src, dst, etype,
                                                         sum2, E, HID, R);
    build_w_kernel<<<cdiv(R * HID * EMBD, 256), 256, 0, stream>>>(comp2, basis2, W2,
                                                                  R, HID * EMBD);
    wmma_gemm_kernel<128, 8, false><<<gemm_blocks(Nn, 128, 8), 256, 0, stream>>>(
        sum2, W2, agg2, Nn, R * HID, nullptr, 0);
    wmma_gemm_kernel<128, 8, false><<<gemm_blocks(Nn, 128, 8), 256, 0, stream>>>(
        h1, root2, root2c, Nn, HID, nullptr, 0);
    rgcn_epilogue_kernel<<<cdiv(Nn * EMBD, 256), 256, 0, stream>>>(
        agg2, root2c, bias2, deg, h2, Nn, EMBD);

    // ---- decoder + softmax -----------------------------------------------
    wmma_gemm_kernel<32, 2, false><<<gemm_blocks(Nn, 32, 2), 256, 0, stream>>>(
        h2, w_dec, logits, Nn, EMBD, nullptr, 0);
    softmax32_kernel<<<cdiv(Nn * 32, 256), 256, 0, stream>>>(logits, (float*)d_out, Nn);
}
